// GlobalTransfomer_80513456931529
// MI455X (gfx1250) — matrix operand order repaired and checked by reading, not compiled, not run on hardware
//
#include <hip/hip_runtime.h>

// ---------------------------------------------------------------------------
// PointTransformer block for MI455X (gfx1250, wave32, WMMA bf16 16x16x32)
// Fragment-order LDS staging: every WMMA operand is a contiguous 32B vector.
// ---------------------------------------------------------------------------

typedef __bf16 bf16;
typedef __attribute__((ext_vector_type(16))) __bf16 v16bf;
typedef __attribute__((ext_vector_type(8)))  float  v8f;

constexpr int Bb = 4;
constexpr int Nn = 8192;
constexpr int Ss = 512;
constexpr int Kk = 16;
constexpr int Ff = 128;
constexpr float EPSf = 1e-5f;

__device__ __forceinline__ v8f wmma_bf16(v16bf a, v16bf b, v8f c) {
  return __builtin_amdgcn_wmma_f32_16x16x32_bf16(false, a, false, b, (short)0, c,
                                                 false, false);
}

// A-fragment element position: (row 0..15, k 0..127) -> index in bf16[2048]
// Loader reads lane L: row=L%16, k = kw*32 + (L>=16?8:0) + (h%8) + (h/8)*16.
__device__ __forceinline__ int afrag_pos(int row, int k) {
  const int kw = k >> 5, kr = k & 31;
  const int hi = (kr >> 3) & 1;
  const int lane = row + (hi << 4);
  const int h = (kr & 7) + ((kr >> 4) << 3);
  return ((kw << 5) + lane) * 16 + h;
}

// GEMM on fragment-order operands: A frag bf16[4][32][16], W frag bf16[4][8][32][16]
__device__ __forceinline__ void gemm_strip_frag(const bf16* Af, const bf16* Wf,
                                                v8f acc[8]) {
  const int lane = threadIdx.x & 31;
#pragma unroll
  for (int kw = 0; kw < 4; ++kw) {
    const v16bf af = *(const v16bf*)(Af + ((kw << 5) + lane) * 16);
#pragma unroll
    for (int nt = 0; nt < 8; ++nt) {
      const v16bf bf = *(const v16bf*)(Wf + ((((kw << 3) + nt) << 5) + lane) * 16);
      acc[nt] = wmma_bf16(af, bf, acc[nt]);
    }
  }
}

// GEMM with A streamed from global fp32 (row-major, 128 cols), W frag in LDS
__device__ __forceinline__ void gemm_strip_gA(const float* A, const bf16* Wf,
                                              v8f acc[8]) {
  const int lane = threadIdx.x & 31;
  const int row  = lane & 15;
  const int hi   = (lane >> 4) & 1;
  const float* ar = A + row * 128;
#pragma unroll
  for (int kw = 0; kw < 4; ++kw) {
    const int kb = kw * 32 + hi * 8;
    const float4 f0 = *(const float4*)(ar + kb);
    const float4 f1 = *(const float4*)(ar + kb + 4);
    const float4 f2 = *(const float4*)(ar + kb + 16);
    const float4 f3 = *(const float4*)(ar + kb + 20);
    v16bf af;
    af[0] = (bf16)f0.x; af[1] = (bf16)f0.y; af[2] = (bf16)f0.z; af[3] = (bf16)f0.w;
    af[4] = (bf16)f1.x; af[5] = (bf16)f1.y; af[6] = (bf16)f1.z; af[7] = (bf16)f1.w;
    af[8] = (bf16)f2.x; af[9] = (bf16)f2.y; af[10] = (bf16)f2.z; af[11] = (bf16)f2.w;
    af[12] = (bf16)f3.x; af[13] = (bf16)f3.y; af[14] = (bf16)f3.z; af[15] = (bf16)f3.w;
#pragma unroll
    for (int nt = 0; nt < 8; ++nt) {
      const v16bf bf = *(const v16bf*)(Wf + ((((kw << 3) + nt) << 5) + lane) * 16);
      acc[nt] = wmma_bf16(af, bf, acc[nt]);
    }
  }
}

// copy a pre-swizzled 32KB weight-fragment image from global into LDS
__device__ __forceinline__ void copy_wfrag(const bf16* __restrict__ g, bf16* s) {
  const uint4* gs = (const uint4*)g;
  uint4* ss = (uint4*)s;
  for (int i = threadIdx.x; i < 2048; i += blockDim.x) ss[i] = gs[i];
}

// ---------------------------------------------------------------------------
// 0) Pre-swizzle a 128x128 fp32 weight (optionally row-offset) into B-fragment
//    order bf16[16384]: idx = ((kw*8+nt)*32+lane)*16+h ; k=kw*32+(lane&16?8:0)+(h%8)+(h/8)*16
// ---------------------------------------------------------------------------
__global__ void prep_frag_kernel(const float* __restrict__ src, bf16* __restrict__ dst,
                                 int row_off) {
  const int i = blockIdx.x * blockDim.x + threadIdx.x;  // 0..16383
  const int h = i & 15;
  const int lane = (i >> 4) & 31;
  const int nt = (i >> 9) & 7;
  const int kw = i >> 12;
  const int n = nt * 16 + (lane & 15);
  const int k = kw * 32 + ((lane & 16) ? 8 : 0) + (h & 7) + ((h >> 3) << 4);
  dst[i] = (bf16)src[(size_t)(k + row_off) * 128 + n];
}

// ---------------------------------------------------------------------------
// 1) LayerNorm: one wave per 128-float row
// ---------------------------------------------------------------------------
__global__ void ln_kernel(const float* __restrict__ x, const float* __restrict__ g,
                          const float* __restrict__ b, float* __restrict__ out) {
  const int wid  = blockIdx.x * (blockDim.x >> 5) + (threadIdx.x >> 5);
  const int lane = threadIdx.x & 31;
  const float4 v = ((const float4*)(x + (size_t)wid * 128))[lane];
  float s = v.x + v.y + v.z + v.w;
#pragma unroll
  for (int o = 16; o; o >>= 1) s += __shfl_xor(s, o);
  const float m = s * (1.0f / 128.0f);
  const float dx = v.x - m, dy = v.y - m, dz = v.z - m, dw = v.w - m;
  float s2 = dx * dx + dy * dy + dz * dz + dw * dw;
#pragma unroll
  for (int o = 16; o; o >>= 1) s2 += __shfl_xor(s2, o);
  const float rstd = rsqrtf(s2 * (1.0f / 128.0f) + EPSf);
  const float4 gg = ((const float4*)g)[lane];
  const float4 bb = ((const float4*)b)[lane];
  float4 o4;
  o4.x = dx * rstd * gg.x + bb.x;
  o4.y = dy * rstd * gg.y + bb.y;
  o4.z = dz * rstd * gg.z + bb.z;
  o4.w = dw * rstd * gg.w + bb.w;
  ((float4*)(out + (size_t)wid * 128))[lane] = o4;
}

// ---------------------------------------------------------------------------
// 2) Farthest point sampling: one 1024-thread block per batch, points in regs
// ---------------------------------------------------------------------------
__global__ void fps_kernel(const float* __restrict__ xyzp, int* __restrict__ fps_idx) {
  const int b = blockIdx.x;
  const int tid = threadIdx.x;
  constexpr int PPT = Nn / 1024;  // 8
  float px[PPT], py[PPT], pz[PPT], md[PPT];
#pragma unroll
  for (int t = 0; t < PPT; ++t) {
    const int i = tid + t * 1024;
    const float* p = xyzp + ((size_t)b * Nn + i) * 4;
    px[t] = p[0]; py[t] = p[1]; pz[t] = p[2]; md[t] = 1e10f;
  }
  __shared__ float lp[3];
  __shared__ float swd[32];
  __shared__ int   swi[32];
  if (tid == 0) {
    fps_idx[b * Ss] = 0;
    const float* p = xyzp + (size_t)b * Nn * 4;
    lp[0] = p[0]; lp[1] = p[1]; lp[2] = p[2];
  }
  __syncthreads();
  for (int s = 1; s < Ss; ++s) {
    const float lx = lp[0], ly = lp[1], lz = lp[2];
    float bd = -1.0f; int bi = 0x7fffffff;
#pragma unroll
    for (int t = 0; t < PPT; ++t) {
      const float dx = px[t] - lx, dy = py[t] - ly, dz = pz[t] - lz;
      md[t] = fminf(md[t], dx * dx + dy * dy + dz * dz);
      const int i = tid + t * 1024;
      if (md[t] > bd) { bd = md[t]; bi = i; }
    }
#pragma unroll
    for (int o = 16; o; o >>= 1) {
      const float od = __shfl_down(bd, o);
      const int   oi = __shfl_down(bi, o);
      if (od > bd || (od == bd && oi < bi)) { bd = od; bi = oi; }
    }
    if ((tid & 31) == 0) { swd[tid >> 5] = bd; swi[tid >> 5] = bi; }
    __syncthreads();
    if (tid < 32) {
      bd = swd[tid]; bi = swi[tid];
#pragma unroll
      for (int o = 16; o; o >>= 1) {
        const float od = __shfl_down(bd, o);
        const int   oi = __shfl_down(bi, o);
        if (od > bd || (od == bd && oi < bi)) { bd = od; bi = oi; }
      }
      if (tid == 0) {
        fps_idx[b * Ss + s] = bi;
        const float* p = xyzp + ((size_t)b * Nn + bi) * 4;
        lp[0] = p[0]; lp[1] = p[1]; lp[2] = p[2];
      }
    }
    __syncthreads();
  }
}

// ---------------------------------------------------------------------------
// 3) KNN (downsample): 512 queries/batch vs 8192 refs, top-16, LDS ref tiles
// ---------------------------------------------------------------------------
__global__ void knn_dn_kernel(const float* __restrict__ xyzp,
                              const int* __restrict__ fps_idx,
                              int* __restrict__ idx_dn, float* __restrict__ new_xyz,
                              float* __restrict__ new_xyzp) {
  const int b = blockIdx.x >> 1;
  const int s = ((blockIdx.x & 1) << 8) + threadIdx.x;
  __shared__ float sref[1024 * 3];
  const int qi = fps_idx[b * Ss + s];
  const float* qp = xyzp + ((size_t)b * Nn + qi) * 4;
  const float qx = qp[0], qy = qp[1], qz = qp[2];
  float bd[Kk]; int bi[Kk];
#pragma unroll
  for (int t = 0; t < Kk; ++t) { bd[t] = 1e30f; bi[t] = 0; }
  for (int tile = 0; tile < Nn; tile += 1024) {
    __syncthreads();
    for (int t = threadIdx.x; t < 1024; t += blockDim.x) {
      const float* p = xyzp + ((size_t)b * Nn + tile + t) * 4;
      sref[t * 3 + 0] = p[0]; sref[t * 3 + 1] = p[1]; sref[t * 3 + 2] = p[2];
    }
    __syncthreads();
    for (int t = 0; t < 1024; ++t) {
      const float dx = qx - sref[t * 3], dy = qy - sref[t * 3 + 1],
                  dz = qz - sref[t * 3 + 2];
      const float d = dx * dx + dy * dy + dz * dz;
      if (d < bd[Kk - 1]) {
        bd[Kk - 1] = d; bi[Kk - 1] = tile + t;
#pragma unroll
        for (int u = Kk - 1; u > 0; --u)
          if (bd[u] < bd[u - 1]) {
            const float td = bd[u]; bd[u] = bd[u - 1]; bd[u - 1] = td;
            const int   ti = bi[u]; bi[u] = bi[u - 1]; bi[u - 1] = ti;
          }
      }
    }
  }
  const int base = (b * Ss + s) * Kk;
#pragma unroll
  for (int t = 0; t < Kk; ++t) idx_dn[base + t] = bi[t];
  new_xyz[(b * Ss + s) * 3 + 0] = qx;
  new_xyz[(b * Ss + s) * 3 + 1] = qy;
  new_xyz[(b * Ss + s) * 3 + 2] = qz;
  const float* np = xyzp + ((size_t)b * Nn + bi[0]) * 4;
#pragma unroll
  for (int d = 0; d < 4; ++d) new_xyzp[(b * Ss + s) * 4 + d] = np[d];
}

// ---------------------------------------------------------------------------
// 4) KNN (upsample): 8192 queries/batch vs 512 refs, top-16
// ---------------------------------------------------------------------------
__global__ void knn_up_kernel(const float* __restrict__ xyzp,
                              const float* __restrict__ new_xyz,
                              int* __restrict__ idx_up) {
  const int b = blockIdx.x >> 5;
  const int n = ((blockIdx.x & 31) << 8) + threadIdx.x;
  __shared__ float sref[Ss * 3];
  for (int t = threadIdx.x; t < Ss * 3; t += blockDim.x)
    sref[t] = new_xyz[b * Ss * 3 + t];
  __syncthreads();
  const float* qp = xyzp + ((size_t)b * Nn + n) * 4;
  const float qx = qp[0], qy = qp[1], qz = qp[2];
  float bd[Kk]; int bi[Kk];
#pragma unroll
  for (int t = 0; t < Kk; ++t) { bd[t] = 1e30f; bi[t] = 0; }
  for (int t = 0; t < Ss; ++t) {
    const float dx = qx - sref[t * 3], dy = qy - sref[t * 3 + 1],
                dz = qz - sref[t * 3 + 2];
    const float d = dx * dx + dy * dy + dz * dz;
    if (d < bd[Kk - 1]) {
      bd[Kk - 1] = d; bi[Kk - 1] = t;
#pragma unroll
      for (int u = Kk - 1; u > 0; --u)
        if (bd[u] < bd[u - 1]) {
          const float td = bd[u]; bd[u] = bd[u - 1]; bd[u - 1] = td;
          const int   ti = bi[u]; bi[u] = bi[u - 1]; bi[u - 1] = ti;
        }
    }
  }
  const size_t base = ((size_t)b * Nn + n) * Kk;
#pragma unroll
  for (int t = 0; t < Kk; ++t) idx_up[base + t] = bi[t];
}

// ---------------------------------------------------------------------------
// 5) Grouped MLP0: h0 = [grouped_xyz | grouped_feat] @ td_w0 + td_b0 (pre-BN)
//    one wave per (b,s) group; 16 neighbor rows = WMMA M dimension
// ---------------------------------------------------------------------------
__global__ void td0_kernel(const float* __restrict__ xyzp,
                           const float* __restrict__ feats,
                           const int* __restrict__ idx_dn,
                           const float* __restrict__ new_xyz,
                           const bf16* __restrict__ w0f,   // pre-swizzled rows 3..130
                           const float* __restrict__ w0,   // raw (for first 3 rows)
                           const float* __restrict__ b0, float* __restrict__ h0) {
  __shared__ __align__(32) bf16 sw[16384];
  __shared__ float sw3[3 * 128];
  __shared__ float sb[128];
  __shared__ __align__(32) bf16 sA[2][2048];
  __shared__ float sX[2][16 * 3];
  __shared__ int   sI[2][16];
  copy_wfrag(w0f, sw);
  for (int i = threadIdx.x; i < 3 * 128; i += blockDim.x) sw3[i] = w0[i];
  for (int i = threadIdx.x; i < 128; i += blockDim.x) sb[i] = b0[i];
  const int w = threadIdx.x >> 5, lane = threadIdx.x & 31;
  const int g = blockIdx.x * 2 + w;     // (b,s) flat, 0..2047
  const int b = g >> 9;
  __syncthreads();
  if (lane < 16) {
    const int ii = idx_dn[g * Kk + lane];
    sI[w][lane] = ii;
    const float* p = xyzp + ((size_t)b * Nn + ii) * 4;
    sX[w][lane * 3 + 0] = p[0] - new_xyz[g * 3 + 0];
    sX[w][lane * 3 + 1] = p[1] - new_xyz[g * 3 + 1];
    sX[w][lane * 3 + 2] = p[2] - new_xyz[g * 3 + 2];
  }
  __syncthreads();
  for (int j = 0; j < 16; ++j) {
    const float* fr = feats + ((size_t)b * Nn + sI[w][j]) * 128;
    for (int c = lane; c < 128; c += 32) sA[w][afrag_pos(j, c)] = (bf16)fr[c];
  }
  __syncthreads();
  v8f acc[8] = {};
  gemm_strip_frag(sA[w], sw, acc);
  const int col16 = lane & 15, mb = (lane < 16) ? 0 : 8;
#pragma unroll
  for (int nt = 0; nt < 8; ++nt) {
    const int c = nt * 16 + col16;
#pragma unroll
    for (int r = 0; r < 8; ++r) {
      const int row = mb + r;
      const float v = acc[nt][r] + sb[c] + sX[w][row * 3 + 0] * sw3[c] +
                      sX[w][row * 3 + 1] * sw3[128 + c] +
                      sX[w][row * 3 + 2] * sw3[256 + c];
      h0[((size_t)g * 16 + row) * 128 + c] = v;
    }
  }
}

// ---------------------------------------------------------------------------
// BatchNorm stats: coalesced partial sums + atomics, then fold into scale/shift
// ---------------------------------------------------------------------------
__global__ void zero_kernel(float* __restrict__ p, int n) {
  const int i = blockIdx.x * blockDim.x + threadIdx.x;
  if (i < n) p[i] = 0.0f;
}

__global__ void bn_sum_kernel(const float* __restrict__ src, float* __restrict__ sums,
                              int rowsPerBlock) {
  const int c = threadIdx.x;  // 128 channels
  const size_t base = (size_t)blockIdx.x * rowsPerBlock * 128;
  float s = 0.0f, s2 = 0.0f;
  for (int r = 0; r < rowsPerBlock; ++r) {
    const float v = src[base + (size_t)r * 128 + c];
    s += v; s2 += v * v;
  }
  atomicAdd(&sums[c], s);
  atomicAdd(&sums[128 + c], s2);
}

__global__ void bn_fin_kernel(const float* __restrict__ sums,
                              const float* __restrict__ g, const float* __restrict__ b,
                              float invM, float* __restrict__ scale,
                              float* __restrict__ shift) {
  const int c = threadIdx.x;
  const float m   = sums[c] * invM;
  const float var = sums[128 + c] * invM - m * m;
  const float sc  = rsqrtf(var + EPSf) * g[c];
  scale[c] = sc;
  shift[c] = b[c] - m * sc;
}

// ---------------------------------------------------------------------------
// 7) Grouped MLP1: h1 = relu(bn0(h0)) @ td_w1 + td_b1 (pre-BN)
// ---------------------------------------------------------------------------
__global__ void td1_kernel(const float* __restrict__ h0, const bf16* __restrict__ w1f,
                           const float* __restrict__ b1,
                           const float* __restrict__ scale0,
                           const float* __restrict__ shift0, float* __restrict__ h1) {
  __shared__ __align__(32) bf16 sw[16384];
  __shared__ float sb[128], ssc[128], ssh[128];
  __shared__ __align__(32) bf16 sA[2][2048];
  copy_wfrag(w1f, sw);
  for (int i = threadIdx.x; i < 128; i += blockDim.x) {
    sb[i] = b1[i]; ssc[i] = scale0[i]; ssh[i] = shift0[i];
  }
  const int w = threadIdx.x >> 5, lane = threadIdx.x & 31;
  const int g = blockIdx.x * 2 + w;
  __syncthreads();
  for (int t = lane; t < 16 * 128; t += 32) {
    const int c = t & 127;
    const float v = h0[(size_t)g * 2048 + t] * ssc[c] + ssh[c];
    sA[w][afrag_pos(t >> 7, c)] = (bf16)fmaxf(v, 0.0f);
  }
  __syncthreads();
  v8f acc[8] = {};
  gemm_strip_frag(sA[w], sw, acc);
  const int col16 = lane & 15, mb = (lane < 16) ? 0 : 8;
#pragma unroll
  for (int nt = 0; nt < 8; ++nt) {
    const int c = nt * 16 + col16;
#pragma unroll
    for (int r = 0; r < 8; ++r)
      h1[((size_t)g * 16 + mb + r) * 128 + c] = acc[nt][r] + sb[c];
  }
}

// ---------------------------------------------------------------------------
// 9) BN1 + relu + max over the K neighbors -> new_feats (B,S,128)
// ---------------------------------------------------------------------------
__global__ void tdmax_kernel(const float* __restrict__ h1,
                             const float* __restrict__ scale1,
                             const float* __restrict__ shift1,
                             float* __restrict__ new_feats) {
  const int i = blockIdx.x * blockDim.x + threadIdx.x;  // over B*S*128
  const int c = i & 127, g = i >> 7;
  const float sc = scale1[c], sh = shift1[c];
  float m = -1e30f;
#pragma unroll
  for (int j = 0; j < Kk; ++j) {
    const float v = fmaxf(h1[(size_t)g * 2048 + j * 128 + c] * sc + sh, 0.0f);
    m = fmaxf(m, v);
  }
  new_feats[i] = m;
}

// ---------------------------------------------------------------------------
// Generic WMMA GEMM: out = A(rows x 128) @ W(128x128) + bias (+ addsrc)
// 4 waves/block, one wave per 16-row strip; W pre-swizzled bf16 fragments
// ---------------------------------------------------------------------------
__global__ void gemm128_kernel(const float* __restrict__ A, const bf16* __restrict__ Wf,
                               const float* __restrict__ bias,
                               const float* __restrict__ addsrc,
                               float* __restrict__ out) {
  __shared__ __align__(32) bf16 sw[16384];
  __shared__ float sb[128];
  const int wid = blockIdx.x * (blockDim.x >> 5) + (threadIdx.x >> 5);
  const size_t r0 = (size_t)wid * 16;
  __builtin_prefetch(A + r0 * 128, 0, 1);   // global_prefetch_b8
  copy_wfrag(Wf, sw);
  for (int i = threadIdx.x; i < 128; i += blockDim.x)
    sb[i] = bias ? bias[i] : 0.0f;
  __syncthreads();
  v8f acc[8] = {};
  gemm_strip_gA(A + r0 * 128, sw, acc);
  const int lane = threadIdx.x & 31;
  const int col16 = lane & 15, mb = (lane < 16) ? 0 : 8;
#pragma unroll
  for (int nt = 0; nt < 8; ++nt) {
    const int c = nt * 16 + col16;
#pragma unroll
    for (int r = 0; r < 8; ++r) {
      const size_t row = r0 + mb + r;
      float v = acc[nt][r] + sb[c];
      if (addsrc) v += addsrc[row * 128 + c];
      out[row * 128 + c] = v;
    }
  }
}

// ---------------------------------------------------------------------------
// Fused attention: one wave per point (b,n); pos_enc never hits HBM.
//   pe  = relu(pdiff@pe_w1+b1)@pe_w2+b2
//   a   = relu((q-k+pe)@at_w1+b1)@at_w2+b2 ; softmax over K (M-dim)
//   res = sum_K softmax * (v + pe)
// All inter-GEMM tensors are written straight into A-fragment-order bf16 LDS.
// Softmax over M uses the C/D layout: lanes L and L+16 hold rows 0-7 / 8-15 of
// column L%16 -> per-lane reduce + __shfl_xor(.,16).
// ---------------------------------------------------------------------------
__global__ void attn_kernel(const float* __restrict__ xyzp,
                            const float* __restrict__ new_xyzp,
                            const int* __restrict__ idx_up,
                            const float* __restrict__ qbuf,
                            const float* __restrict__ kmat,
                            const float* __restrict__ vmat,
                            const float* __restrict__ pe_w1,
                            const float* __restrict__ pe_b1,
                            const bf16* __restrict__ pe_w2f,
                            const float* __restrict__ pe_b2,
                            const bf16* __restrict__ at_w1f,
                            const float* __restrict__ at_b1,
                            const bf16* __restrict__ at_w2f,
                            const float* __restrict__ at_b2,
                            float* __restrict__ res) {
  __shared__ __align__(32) bf16 sW2[16384];
  __shared__ __align__(32) bf16 sA1[16384];
  __shared__ __align__(32) bf16 sA2[16384];
  __shared__ float sPW1[4 * 128];
  __shared__ float sB[4 * 128];   // pe_b1 | pe_b2 | at_b1 | at_b2
  __shared__ __align__(32) bf16 sS[2][2048];   // A-fragment scratch
  __shared__ float sPE[2][16 * 128];
  __shared__ int   sI[2][16];
  __shared__ float sPD[2][16 * 4];
  copy_wfrag(pe_w2f, sW2);
  copy_wfrag(at_w1f, sA1);
  copy_wfrag(at_w2f, sA2);
  for (int i = threadIdx.x; i < 4 * 128; i += blockDim.x) sPW1[i] = pe_w1[i];
  for (int i = threadIdx.x; i < 128; i += blockDim.x) {
    sB[i] = pe_b1[i]; sB[128 + i] = pe_b2[i];
    sB[256 + i] = at_b1[i]; sB[384 + i] = at_b2[i];
  }
  const int w = threadIdx.x >> 5, lane = threadIdx.x & 31;
  const int p = blockIdx.x * 2 + w;   // flat (b,n), 0..32767
  const int b = p >> 13;
  __syncthreads();
  if (lane < 16) {
    const int ii = idx_up[(size_t)p * Kk + lane];
    sI[w][lane] = ii;
    const float* qq = xyzp + (size_t)p * 4;
    const float* rr = new_xyzp + ((size_t)(b * Ss + ii)) * 4;
#pragma unroll
    for (int d = 0; d < 4; ++d) sPD[w][lane * 4 + d] = qq[d] - rr[d];
  }
  __syncthreads();
  // pe1 = relu(pdiff @ pe_w1 + pe_b1) -> A-frag
  for (int t = lane; t < 16 * 128; t += 32) {
    const int j = t >> 7, c = t & 127;
    const float* pd = &sPD[w][j * 4];
    const float v = sB[c] + pd[0] * sPW1[c] + pd[1] * sPW1[128 + c] +
                    pd[2] * sPW1[256 + c] + pd[3] * sPW1[384 + c];
    sS[w][afrag_pos(j, c)] = (bf16)fmaxf(v, 0.0f);
  }
  __syncthreads();
  v8f accPE[8] = {};
  gemm_strip_frag(sS[w], sW2, accPE);
  const int col16 = lane & 15, mb = (lane < 16) ? 0 : 8;
  __syncthreads();
  // pe -> sPE (f32); a = q - k + pe -> A-frag
#pragma unroll
  for (int nt = 0; nt < 8; ++nt) {
    const int c = nt * 16 + col16;
    const float qv = qbuf[(size_t)p * 128 + c];
#pragma unroll
    for (int r = 0; r < 8; ++r) {
      const int row = mb + r;
      const float pe = accPE[nt][r] + sB[128 + c];
      sPE[w][row * 128 + c] = pe;
      const float kv = kmat[((size_t)(b * Ss + sI[w][row])) * 128 + c];
      sS[w][afrag_pos(row, c)] = (bf16)(qv - kv + pe);
    }
  }
  __syncthreads();
  v8f acc1[8] = {};
  gemm_strip_frag(sS[w], sA1, acc1);
  __syncthreads();
#pragma unroll
  for (int nt = 0; nt < 8; ++nt) {
    const int c = nt * 16 + col16;
#pragma unroll
    for (int r = 0; r < 8; ++r)
      sS[w][afrag_pos(mb + r, c)] = (bf16)fmaxf(acc1[nt][r] + sB[256 + c], 0.0f);
  }
  __syncthreads();
  v8f acc2[8] = {};
  gemm_strip_frag(sS[w], sA2, acc2);
  // softmax over the 16 neighbor rows (M dim), per column, then weighted sum
  const float sc = 0.0883883476483184f;  // 1/sqrt(128)
#pragma unroll
  for (int nt = 0; nt < 8; ++nt) {
    const int c = nt * 16 + col16;
    float vals[8];
    float m = -1e30f;
#pragma unroll
    for (int r = 0; r < 8; ++r) {
      vals[r] = acc2[nt][r] + sB[384 + c];
      m = fmaxf(m, vals[r]);
    }
    m = fmaxf(m, __shfl_xor(m, 16));
    float ssum = 0.0f;
#pragma unroll
    for (int r = 0; r < 8; ++r) {
      vals[r] = __expf((vals[r] - m) * sc);
      ssum += vals[r];
    }
    ssum += __shfl_xor(ssum, 16);
    const float inv = 1.0f / ssum;
    float o = 0.0f;
#pragma unroll
    for (int r = 0; r < 8; ++r) {
      const int row = mb + r;
      const float vv = vmat[((size_t)(b * Ss + sI[w][row])) * 128 + c] +
                       sPE[w][row * 128 + c];
      o += vals[r] * inv * vv;
    }
    o += __shfl_xor(o, 16);
    if (lane < 16) res[(size_t)p * 128 + c] = o;
  }
}

// ---------------------------------------------------------------------------
// Host launcher
// ---------------------------------------------------------------------------
extern "C" void kernel_launch(void* const* d_in, const int* in_sizes, int n_in,
                              void* d_out, int out_size, void* d_ws, size_t ws_size,
                              hipStream_t stream) {
  (void)in_sizes; (void)n_in; (void)out_size; (void)ws_size;
  const float* xyzp     = (const float*)d_in[0];
  const float* features = (const float*)d_in[1];
  const float* ln_g     = (const float*)d_in[2];
  const float* ln_b     = (const float*)d_in[3];
  const float* td_w0    = (const float*)d_in[4];
  const float* td_b0    = (const float*)d_in[5];
  const float* td_bn0_g = (const float*)d_in[6];
  const float* td_bn0_b = (const float*)d_in[7];
  const float* td_w1    = (const float*)d_in[8];
  const float* td_b1    = (const float*)d_in[9];
  const float* td_bn1_g = (const float*)d_in[10];
  const float* td_bn1_b = (const float*)d_in[11];
  const float* w_kern   = (const float*)d_in[12];
  const float* b_kern   = (const float*)d_in[13];
  const float* w_gkern  = (const float*)d_in[14];
  const float* b_gkern  = (const float*)d_in[15];
  const float* w_agg    = (const float*)d_in[16];
  const float* b_agg    = (const float*)d_in[17];
  const float* w_q      = (const float*)d_in[18];
  const float* w_k      = (const float*)d_in[19];
  const float* w_v      = (const float*)d_in[20];
  const float* pe_w1    = (const float*)d_in[21];
  const float* pe_b1    = (const float*)d_in[22];
  const float* pe_w2    = (const float*)d_in[23];
  const float* pe_b2    = (const float*)d_in[24];
  const float* at_w1    = (const float*)d_in[25];
  const float* at_b1    = (const float*)d_in[26];
  const float* at_w2    = (const float*)d_in[27];
  const float* at_b2    = (const float*)d_in[28];

  // workspace layout (floats)
  constexpr size_t BIG = (size_t)Bb * Nn * Ff;  // 4194304
  float* ws       = (float*)d_ws;
  float* feats    = ws;               // LN output
  float* qbuf     = ws + BIG;
  float* xres     = ws + 2 * BIG;     // x, later reused for attention result
  float* h0       = ws + 3 * BIG;
  float* h1       = ws + 4 * BIG;
  float* more     = ws + 5 * BIG;
  int*   idx_up   = (int*)more;                         // 524288
  int*   fps_i    = (int*)(more + 524288);              // 2048
  int*   idx_dn   = (int*)(more + 526336);              // 32768
  float* new_xyz  = more + 559104;                      // 6144
  float* new_xyzp = more + 565248;                      // 8192
  float* new_ft   = more + 573440;                      // 262144
  float* new_x    = more + 835584;                      // 262144
  float* kmat     = more + 1097728;                     // 262144
  float* vmat     = more + 1359872;                     // 262144
  float* bnsum    = more + 1622016;                     // 256
  float* scale0   = more + 1622272;
  float* shift0   = more + 1622400;
  float* scale1   = more + 1622528;
  float* shift1   = more + 1622656;                     // +128 -> 1622784
  bf16*  wfrag    = (bf16*)(more + 1622784);            // 11 x 16384 bf16

  bf16* f_td0  = wfrag + 0  * 16384;
  bf16* f_td1  = wfrag + 1  * 16384;
  bf16* f_kern = wfrag + 2  * 16384;
  bf16* f_q    = wfrag + 3  * 16384;
  bf16* f_gk   = wfrag + 4  * 16384;
  bf16* f_k    = wfrag + 5  * 16384;
  bf16* f_v    = wfrag + 6  * 16384;
  bf16* f_agg  = wfrag + 7  * 16384;
  bf16* f_pew2 = wfrag + 8  * 16384;
  bf16* f_atw1 = wfrag + 9  * 16384;
  bf16* f_atw2 = wfrag + 10 * 16384;

  const float invM = 1.0f / (float)(Bb * Ss * Kk);

  // pre-swizzle all 128x128 weights into WMMA B-fragment order (bf16)
  prep_frag_kernel<<<64, 256, 0, stream>>>(td_w0, f_td0, 3);  // rows 3..130
  prep_frag_kernel<<<64, 256, 0, stream>>>(td_w1, f_td1, 0);
  prep_frag_kernel<<<64, 256, 0, stream>>>(w_kern, f_kern, 0);
  prep_frag_kernel<<<64, 256, 0, stream>>>(w_q, f_q, 0);
  prep_frag_kernel<<<64, 256, 0, stream>>>(w_gkern, f_gk, 0);
  prep_frag_kernel<<<64, 256, 0, stream>>>(w_k, f_k, 0);
  prep_frag_kernel<<<64, 256, 0, stream>>>(w_v, f_v, 0);
  prep_frag_kernel<<<64, 256, 0, stream>>>(w_agg, f_agg, 0);
  prep_frag_kernel<<<64, 256, 0, stream>>>(pe_w2, f_pew2, 0);
  prep_frag_kernel<<<64, 256, 0, stream>>>(at_w1, f_atw1, 0);
  prep_frag_kernel<<<64, 256, 0, stream>>>(at_w2, f_atw2, 0);

  ln_kernel<<<4096, 256, 0, stream>>>(features, ln_g, ln_b, feats);
  fps_kernel<<<Bb, 1024, 0, stream>>>(xyzp, fps_i);
  knn_dn_kernel<<<8, 256, 0, stream>>>(xyzp, fps_i, idx_dn, new_xyz, new_xyzp);
  knn_up_kernel<<<128, 256, 0, stream>>>(xyzp, new_xyz, idx_up);

  td0_kernel<<<1024, 64, 0, stream>>>(xyzp, feats, idx_dn, new_xyz, f_td0, td_w0,
                                      td_b0, h0);
  zero_kernel<<<1, 256, 0, stream>>>(bnsum, 256);
  bn_sum_kernel<<<128, 128, 0, stream>>>(h0, bnsum, 256);
  bn_fin_kernel<<<1, 128, 0, stream>>>(bnsum, td_bn0_g, td_bn0_b, invM, scale0, shift0);
  td1_kernel<<<1024, 64, 0, stream>>>(h0, f_td1, td_b1, scale0, shift0, h1);
  zero_kernel<<<1, 256, 0, stream>>>(bnsum, 256);
  bn_sum_kernel<<<128, 128, 0, stream>>>(h1, bnsum, 256);
  bn_fin_kernel<<<1, 128, 0, stream>>>(bnsum, td_bn1_g, td_bn1_b, invM, scale1, shift1);
  tdmax_kernel<<<1024, 256, 0, stream>>>(h1, scale1, shift1, new_ft);

  gemm128_kernel<<<512, 128, 0, stream>>>(feats, f_kern, b_kern, nullptr, xres);  // x
  gemm128_kernel<<<512, 128, 0, stream>>>(xres, f_q, nullptr, nullptr, qbuf);     // q
  gemm128_kernel<<<32, 128, 0, stream>>>(new_ft, f_gk, b_gkern, nullptr, new_x);
  gemm128_kernel<<<32, 128, 0, stream>>>(new_x, f_k, nullptr, nullptr, kmat);
  gemm128_kernel<<<32, 128, 0, stream>>>(new_x, f_v, nullptr, nullptr, vmat);

  attn_kernel<<<16384, 64, 0, stream>>>(xyzp, new_xyzp, idx_up, qbuf, kmat, vmat,
                                        pe_w1, pe_b1, f_pew2, pe_b2, f_atw1, at_b1,
                                        f_atw2, at_b2, xres);

  gemm128_kernel<<<512, 128, 0, stream>>>(xres, f_agg, b_agg, feats, (float*)d_out);
}
